// LBNER_81088982548713
// MI455X (gfx1250) — matrix operand order repaired and checked by reading, not compiled, not run on hardware
//
#include <hip/hip_runtime.h>
#include <hip/hip_bf16.h>
#include <cstdint>

// ---------------------------------------------------------------------------
// Problem constants (from reference): B=8, L=512, D=768, windows 3/5/7, 9 labels
// ---------------------------------------------------------------------------
#define BB     8
#define LSEQ   512
#define DD     768
#define FOURD  3072
#define NLAB   9
#define NROWS  (BB * LSEQ)   // 4096

typedef __attribute__((ext_vector_type(16))) __bf16 v16bf;
typedef __attribute__((ext_vector_type(8)))  float  v8f;

union FragAB { uint4 q[2]; v16bf v; };   // 32 bytes: 16 bf16 per lane
union FragC  { float f[8]; v8f  v; };    // 32 bytes: 8 f32 per lane

__device__ __forceinline__ uint16_t f2bf(float f) {
    uint32_t u = __float_as_uint(f);
    uint32_t r = u + 0x7FFFu + ((u >> 16) & 1u);   // round-to-nearest-even
    return (uint16_t)(r >> 16);
}
__device__ __forceinline__ float bf2f(uint16_t h) {
    return __uint_as_float(((uint32_t)h) << 16);
}
__device__ __forceinline__ float sigf(float x) {
    return 1.0f / (1.0f + __expf(-x));
}

// CDNA5 async global->LDS copy (ASYNCcnt path, CDNA5 ISA)
__device__ __forceinline__ void async_load_b128(uint32_t lds_off,
                                                const void* gaddr) {
    asm volatile("global_load_async_to_lds_b128 %0, %1, off"
                 :: "v"(lds_off), "v"(gaddr) : "memory");
}
__device__ __forceinline__ void wait_async0() {
    asm volatile("s_wait_asynccnt 0x0" ::: "memory");
}

// ---------------------------------------------------------------------------
// f32 -> bf16 conversion (grid-stride)
// ---------------------------------------------------------------------------
__global__ void cvt_bf16_kernel(const float* __restrict__ src,
                                uint16_t* __restrict__ dst, int n) {
    for (int i = blockIdx.x * blockDim.x + threadIdx.x; i < n;
         i += gridDim.x * blockDim.x)
        dst[i] = f2bf(src[i]);
}

// b_ih + b_hh (added every step in the reference)
__global__ void bias_sum_kernel(const float* __restrict__ bih,
                                const float* __restrict__ bhh,
                                float* __restrict__ bsum, int n) {
    int i = blockIdx.x * blockDim.x + threadIdx.x;
    if (i < n) bsum[i] = bih[i] + bhh[i];
}

// ---------------------------------------------------------------------------
// Input-projection GEMM: gin[k][row][n] = x[row] . Wih[k][n] + (b_ih+b_hh)[k][n]
// The windowed LSTM inputs are shifted views of x, so this is computed ONCE
// per window instead of once per step (15 matmuls -> 3).
// grid = (128 row tiles of 32, 3 windows); block = 1024 threads = 32 wave32s.
// Waves 0..15 handle rows 0..15 of the tile, waves 16..31 rows 16..31; the two
// row groups read identical weight fragments (served by WGP$).
// gin is stored bf16 (gates re-accumulate nothing afterwards; bias folded in).
// ---------------------------------------------------------------------------
__global__ __launch_bounds__(1024)
void gin_kernel(const uint16_t* __restrict__ xb,    // [B*L][D] bf16
                const uint16_t* __restrict__ wihb,  // [3][4D][D] bf16
                const float*    __restrict__ bsum,  // [3][4D]
                uint16_t*       __restrict__ gin)   // [3][B*L][4D] bf16
{
    __shared__ alignas(16) uint16_t xA[32][DD];   // 48 KB

    const int k    = blockIdx.y;
    const int b    = blockIdx.x >> 4;          // 16 tiles of 32 rows per batch
    const int l0   = (blockIdx.x & 15) << 5;

    const int tid  = threadIdx.x;
    const int lane = tid & 31;
    const int wave = tid >> 5;                 // 0..31
    const int rg   = wave >> 4;                // row group 0/1
    const int wv   = wave & 15;                // wave within row group
    const int col  = lane & 15;
    const int half = lane >> 4;
    const int u0   = wv * 48;

    // ---- async global->LDS load of the 32x768 x tile ----
    {
        const int r   = tid >> 5;              // row 0..31, 32 threads per row
        const int l32 = tid & 31;              // 24 bf16 = 48 B each
        const uint16_t* src = xb + ((size_t)b * LSEQ + l0 + r) * DD + l32 * 24;
        const uint32_t  dst = (uint32_t)(uintptr_t)(&xA[r][l32 * 24]);
        async_load_b128(dst,      src);
        async_load_b128(dst + 16, src + 8);
        async_load_b128(dst + 32, src + 16);
    }
    wait_async0();
    __syncthreads();

    FragC acc[4][3];
    #pragma unroll
    for (int g = 0; g < 4; ++g)
        #pragma unroll
        for (int s = 0; s < 3; ++s) {
            const float bv = bsum[k * FOURD + g * DD + u0 + s * 16 + col];
            #pragma unroll
            for (int v = 0; v < 8; ++v) acc[g][s].f[v] = bv;
        }

    const size_t wkbase = (size_t)k * FOURD * DD;
    const int    arow   = rg * 16 + col;

    #pragma unroll 1
    for (int kb = 0; kb < DD; kb += 32) {
        FragAB ax;
        ax.q[0] = *(const uint4*)&xA[arow][kb + half * 8];
        ax.q[1] = *(const uint4*)&xA[arow][kb + 16 + half * 8];

        #pragma unroll
        for (int i = 0; i < 12; ++i) {
            const int g = i / 3, s = i % 3;
            const int n0 = g * DD + u0 + s * 16;
            const uint16_t* wp =
                wihb + wkbase + (size_t)(n0 + col) * DD + kb + half * 8;
            FragAB bw;
            bw.q[0] = *(const uint4*)wp;
            bw.q[1] = *(const uint4*)(wp + 16);
            acc[g][s].v = __builtin_amdgcn_wmma_f32_16x16x32_bf16(
                false, ax.v, false, bw.v, (short)0, acc[g][s].v, false, false);
        }
    }

    // ---- store gates (bf16, bias included) ----
    #pragma unroll
    for (int g = 0; g < 4; ++g)
        #pragma unroll
        for (int s = 0; s < 3; ++s) {
            const int n0 = g * DD + u0 + s * 16;
            #pragma unroll
            for (int v = 0; v < 8; ++v) {
                const int row = rg * 16 + v + (half << 3);
                gin[((size_t)k * NROWS + (size_t)b * LSEQ + l0 + row) * FOURD
                    + n0 + col] = f2bf(acc[g][s].f[v]);
            }
        }
}

// ---------------------------------------------------------------------------
// Recurrent kernel: per step, gates = gin[clamped window index] + h @ Whh^T.
// t = 0 needs no matmul (h = 0). grid = (128 row tiles of 32, 3 windows);
// block = 1024 threads = 32 wave32s (same wave mapping as gin_kernel).
// h lives in LDS (bf16, WMMA A operand), c lives in VGPRs in WMMA C layout.
// Validity masking == jnp.where(mask, new, old): masked rows keep state.
// ---------------------------------------------------------------------------
__global__ __launch_bounds__(1024)
void lstm_rec_kernel(const uint16_t* __restrict__ gin,   // [3][B*L][4D] bf16
                     const uint16_t* __restrict__ whhb,  // [3][4D][D] bf16
                     uint16_t*       __restrict__ hfin)  // [3][B*L][D] bf16
{
    __shared__ alignas(16) uint16_t hA[32][DD];   // 48 KB

    const int k    = blockIdx.y;
    const int w    = 3 + 2 * k;
    const int hw   = k + 1;
    const int b    = blockIdx.x >> 4;
    const int l0   = (blockIdx.x & 15) << 5;

    const int tid  = threadIdx.x;
    const int lane = tid & 31;
    const int wave = tid >> 5;
    const int rg   = wave >> 4;
    const int wv   = wave & 15;
    const int col  = lane & 15;
    const int half = lane >> 4;
    const int u0   = wv * 48;

    for (int i = tid; i < 32 * DD; i += 1024) (&hA[0][0])[i] = 0;

    FragC c[3];
    #pragma unroll
    for (int s = 0; s < 3; ++s)
        #pragma unroll
        for (int v = 0; v < 8; ++v) c[s].f[v] = 0.0f;

    __syncthreads();

    const size_t wkbase = (size_t)k * FOURD * DD;
    const size_t gbase  = ((size_t)k * NROWS + (size_t)b * LSEQ) * FOURD;
    const int    arow   = rg * 16 + col;

    for (int t = 0; t < w; ++t) {
        // ---- init gates from precomputed input projection (clamped idx) ----
        FragC acc[4][3];
        #pragma unroll
        for (int g = 0; g < 4; ++g)
            #pragma unroll
            for (int s = 0; s < 3; ++s)
                #pragma unroll
                for (int v = 0; v < 8; ++v) {
                    const int row = rg * 16 + v + (half << 3);
                    int gl = l0 + row + t - hw;
                    gl = gl < 0 ? 0 : (gl > LSEQ - 1 ? LSEQ - 1 : gl);
                    acc[g][s].f[v] =
                        bf2f(gin[gbase + (size_t)gl * FOURD
                                 + g * DD + u0 + s * 16 + col]);
                }

        // ---- h @ Whh^T (skip at t==0: h is zero) -- branch-free inner loop --
        if (t > 0) {
            #pragma unroll 1
            for (int kb = 0; kb < DD; kb += 32) {
                FragAB ah;
                ah.q[0] = *(const uint4*)&hA[arow][kb + half * 8];
                ah.q[1] = *(const uint4*)&hA[arow][kb + 16 + half * 8];

                #pragma unroll
                for (int i = 0; i < 12; ++i) {
                    const int g = i / 3, s = i % 3;
                    const int n0 = g * DD + u0 + s * 16;
                    const uint16_t* wp =
                        whhb + wkbase + (size_t)(n0 + col) * DD + kb + half * 8;
                    FragAB bh;
                    bh.q[0] = *(const uint4*)wp;
                    bh.q[1] = *(const uint4*)(wp + 16);
                    acc[g][s].v = __builtin_amdgcn_wmma_f32_16x16x32_bf16(
                        false, ah.v, false, bh.v, (short)0,
                        acc[g][s].v, false, false);
                }
            }
        }

        __syncthreads();   // all waves done reading hA before it is rewritten

        // ---- activations + masked state update ----
        #pragma unroll
        for (int s = 0; s < 3; ++s)
            #pragma unroll
            for (int v = 0; v < 8; ++v) {
                const int row = rg * 16 + v + (half << 3);
                const int gl  = l0 + row + t - hw;
                const bool valid = (gl >= 0) && (gl < LSEQ);
                const float iv = sigf(acc[0][s].f[v]);
                const float fv = sigf(acc[1][s].f[v]);
                const float gv = tanhf(acc[2][s].f[v]);
                const float ov = sigf(acc[3][s].f[v]);
                const float cn = fv * c[s].f[v] + iv * gv;
                const float hn = ov * tanhf(cn);
                if (valid) {                       // masked rows keep state
                    c[s].f[v] = cn;
                    hA[row][u0 + s * 16 + col] = f2bf(hn);
                }
            }
        __syncthreads();   // hA stable before next step's reads
    }

    // ---- write final h (bf16) for the attention head ----
    {
        const int r   = tid >> 5;                  // 0..31
        const int l32 = tid & 31;
        uint16_t* dst = hfin +
            ((size_t)k * NROWS + (size_t)b * LSEQ + l0 + r) * DD;
        for (int i = l32; i < DD; i += 32) dst[i] = hA[r][i];
    }
}

// ---------------------------------------------------------------------------
// Attention over the 3 window outputs + 768->9 linear head. Wave32 per row.
// ---------------------------------------------------------------------------
__global__ __launch_bounds__(256)
void head_kernel(const float*    __restrict__ x,      // [B*L][D]
                 const uint16_t* __restrict__ hfin,   // [3][B*L][D] bf16
                 const float*    __restrict__ lin_w,  // [9][D]
                 const float*    __restrict__ lin_b,  // [9]
                 float*          __restrict__ out)    // [B*L][9]
{
    const int gid  = blockIdx.x * blockDim.x + threadIdx.x;
    const int row  = gid >> 5;
    const int lane = threadIdx.x & 31;
    if (row >= NROWS) return;

    const float*    xr = x + (size_t)row * DD;
    const uint16_t* h0 = hfin + ((size_t)0 * NROWS + row) * DD;
    const uint16_t* h1 = hfin + ((size_t)1 * NROWS + row) * DD;
    const uint16_t* h2 = hfin + ((size_t)2 * NROWS + row) * DD;

    float d0 = 0.f, d1 = 0.f, d2 = 0.f;
    for (int d = lane; d < DD; d += 32) {
        const float xv = xr[d];
        d0 += xv * bf2f(h0[d]);
        d1 += xv * bf2f(h1[d]);
        d2 += xv * bf2f(h2[d]);
    }
    #pragma unroll
    for (int off = 16; off; off >>= 1) {
        d0 += __shfl_xor(d0, off, 32);
        d1 += __shfl_xor(d1, off, 32);
        d2 += __shfl_xor(d2, off, 32);
    }
    const float scale = 0.03608439182435161f;   // 1/sqrt(768)
    d0 *= scale; d1 *= scale; d2 *= scale;
    const float m  = fmaxf(d0, fmaxf(d1, d2));
    const float e0 = __expf(d0 - m), e1 = __expf(d1 - m), e2 = __expf(d2 - m);
    const float inv = 1.0f / (e0 + e1 + e2);
    const float a0 = e0 * inv, a1 = e1 * inv, a2 = e2 * inv;

    float logit[NLAB];
    #pragma unroll
    for (int c2 = 0; c2 < NLAB; ++c2) logit[c2] = 0.f;
    for (int d = lane; d < DD; d += 32) {
        const float o = xr[d] + a0 * bf2f(h0[d]) + a1 * bf2f(h1[d])
                              + a2 * bf2f(h2[d]);
        #pragma unroll
        for (int c2 = 0; c2 < NLAB; ++c2)
            logit[c2] += o * lin_w[c2 * DD + d];
    }
    #pragma unroll
    for (int c2 = 0; c2 < NLAB; ++c2) {
        #pragma unroll
        for (int off = 16; off; off >>= 1)
            logit[c2] += __shfl_xor(logit[c2], off, 32);
    }
    if (lane == 0) {
        #pragma unroll
        for (int c2 = 0; c2 < NLAB; ++c2)
            out[(size_t)row * NLAB + c2] = logit[c2] + lin_b[c2];
    }
}

// ---------------------------------------------------------------------------
// Launch
// ---------------------------------------------------------------------------
extern "C" void kernel_launch(void* const* d_in, const int* in_sizes, int n_in,
                              void* d_out, int out_size, void* d_ws, size_t ws_size,
                              hipStream_t stream) {
    (void)in_sizes; (void)n_in; (void)out_size; (void)ws_size;

    const float* x    = (const float*)d_in[0];   // (8,512,768)
    const float* Wih  = (const float*)d_in[1];   // (3,3072,768)
    const float* Whh  = (const float*)d_in[2];   // (3,3072,768)
    const float* bih  = (const float*)d_in[3];   // (3,3072)
    const float* bhh  = (const float*)d_in[4];   // (3,3072)
    const float* lw   = (const float*)d_in[5];   // (9,768)
    const float* lb   = (const float*)d_in[6];   // (9,)
    float*       out  = (float*)d_out;           // (8,512,9)

    // workspace layout (~129 MB total)
    char* ws = (char*)d_ws;
    const size_t XB_B  = (size_t)NROWS * DD * 2;            //   6,291,456
    const size_t WB_B  = (size_t)3 * FOURD * DD * 2;        //  14,155,776
    const size_t BS_B  = (size_t)3 * FOURD * 4;             //      36,864
    const size_t GIN_B = (size_t)3 * NROWS * FOURD * 2;     //  75,497,472
    uint16_t* xb   = (uint16_t*)(ws);
    uint16_t* wihb = (uint16_t*)(ws + XB_B);
    uint16_t* whhb = (uint16_t*)(ws + XB_B + WB_B);
    float*    bsum = (float*)   (ws + XB_B + 2 * WB_B);
    uint16_t* gin  = (uint16_t*)(ws + XB_B + 2 * WB_B + BS_B);
    uint16_t* hfin = (uint16_t*)(ws + XB_B + 2 * WB_B + BS_B + GIN_B);

    const int nX = NROWS * DD;          // 3,145,728
    const int nW = 3 * FOURD * DD;      // 7,077,888
    cvt_bf16_kernel<<<2048, 256, 0, stream>>>(x,   xb,   nX);
    cvt_bf16_kernel<<<4096, 256, 0, stream>>>(Wih, wihb, nW);
    cvt_bf16_kernel<<<4096, 256, 0, stream>>>(Whh, whhb, nW);
    bias_sum_kernel<<<(3 * FOURD + 255) / 256, 256, 0, stream>>>(bih, bhh, bsum,
                                                                 3 * FOURD);

    dim3 grid(NROWS / 32, 3);   // 128 row tiles x 3 windows
    gin_kernel<<<grid, 1024, 0, stream>>>(xb, wihb, bsum, gin);
    lstm_rec_kernel<<<grid, 1024, 0, stream>>>(gin, whhb, hfin);

    head_kernel<<<(NROWS * 32 + 255) / 256, 256, 0, stream>>>(x, hfin, lw, lb,
                                                              out);
}
